// RaiFlowAttention_36498632081802
// MI455X (gfx1250) — matrix operand order repaired and checked by reading, not compiled, not run on hardware
//
#include <hip/hip_runtime.h>
#include <hip/hip_bf16.h>

typedef __attribute__((ext_vector_type(16))) __bf16 v16bf;
typedef __attribute__((ext_vector_type(8)))  __bf16 v8bf;
typedef __attribute__((ext_vector_type(8)))  float  v8f;
typedef __attribute__((ext_vector_type(4)))  int    v4i;

#if defined(__HIP_DEVICE_COMPILE__) && defined(__gfx1250__) && \
    __has_builtin(__builtin_amdgcn_global_load_async_to_lds_b128)
#define HAVE_ASYNC_LDS 1
#else
#define HAVE_ASYNC_LDS 0
#endif

#if HAVE_ASYNC_LDS
typedef __attribute__((address_space(1))) v4i as1_v4i;
typedef __attribute__((address_space(3))) v4i as3_v4i;
#endif

// ---------------- CDNA5 helpers ----------------

// 16-byte global -> LDS copy. Uses the gfx1250 async-to-LDS path when the
// toolchain exposes it (ASYNCcnt-tracked, no VGPR data movement), else a
// plain load/store pair.
__device__ __forceinline__ void g2l_16B(const __bf16* g, __bf16* l) {
#if HAVE_ASYNC_LDS
  __builtin_amdgcn_global_load_async_to_lds_b128(
      (as1_v4i*)(void*)g, (as3_v4i*)(void*)l, 0, 0);
#else
  *(v8bf*)l = *(const v8bf*)g;
#endif
}

__device__ __forceinline__ void async_join() {
#if HAVE_ASYNC_LDS
  asm volatile("s_wait_asynccnt 0x0" ::: "memory");
#endif
}

__device__ __forceinline__ v8f wmma_bf16(v16bf a, v16bf b, v8f c) {
  // 8 args: (neg_a, A, neg_b, B, c_mod, C, reuse_a, reuse_b)
  return __builtin_amdgcn_wmma_f32_16x16x32_bf16(
      false, a, false, b, (short)0, c, false, false);
}

// A fragment 16x32 bf16: lane (<16) holds row M=lane, K in {0..7,16..23};
// lane (>=16) holds row M=lane-16, K in {8..15,24..31}.
__device__ __forceinline__ v16bf load_a_frag(const __bf16* row, int kk, int lane) {
  const int k0 = kk + ((lane & 16) ? 8 : 0);
  v8bf lo = *(const v8bf*)(row + k0);
  v8bf hi = *(const v8bf*)(row + k0 + 16);
  v16bf a;
#pragma unroll
  for (int i = 0; i < 8; ++i) { a[i] = lo[i]; a[i + 8] = hi[i]; }
  return a;
}

// B fragment 32x16 bf16: lane holds column N=lane&15, K in {0..7,16..23}
// (lane<16) or {8..15,24..31} (lane>=16). 'row' points at B^T row for this lane's N.
__device__ __forceinline__ v16bf load_b_frag(const __bf16* row, int kk, int lane) {
  const int k0 = kk + ((lane & 16) ? 8 : 0);
  v8bf lo = *(const v8bf*)(row + k0);
  v8bf hi = *(const v8bf*)(row + k0 + 16);
  v16bf b;
#pragma unroll
  for (int i = 0; i < 8; ++i) { b[i] = lo[i]; b[i + 8] = hi[i]; }
  return b;
}

// ---------------- fp32 -> bf16 conversion ----------------

__global__ __launch_bounds__(256) void cvt_f32_bf16(const float* __restrict__ s,
                                                    __bf16* __restrict__ d, int n) {
  int i = blockIdx.x * blockDim.x + threadIdx.x;
  const int stride = gridDim.x * blockDim.x;
  for (; i < n; i += stride) d[i] = (__bf16)s[i];
}

// ---------------- GEMM: C = A(bf16,MxK) * B(bf16,KxN), f32 accum ----------------
// Tile 128x128, BK=32, double-buffered LDS, 8 waves as 2(m) x 4(n):
// each wave 64x32 = 4x2 WMMA tiles.
// MODE 0: qkv+gate epilogue (clamp+RMSNorm q/k, pack v, sigmoid g)
// MODE 1: plain fp32 store to Cout

template <int MODE>
__global__ __launch_bounds__(256) void gemm_bf16_kernel(
    const __bf16* __restrict__ A, const __bf16* __restrict__ B,
    int Kdim, int Ndim,
    __bf16* __restrict__ Qo, __bf16* __restrict__ Ko,
    __bf16* __restrict__ Vo, __bf16* __restrict__ Go,
    float* __restrict__ Cout) {
  __shared__ __bf16 As[2][128][32];   // A tile, row-major (double buffered)
  __shared__ __bf16 Bts[2][128][32];  // B tile transposed: [n][k]
  __shared__ float  red[128][4];      // per-row partial sums (MODE 0)

  const int tid = threadIdx.x;
  const int lane = tid & 31;
  const int wave = tid >> 5;
  const int wm = wave >> 2;   // 0..1
  const int wn = wave & 3;    // 0..3
  const int lx = lane & 15;
  const int hf = (lane >> 4) & 1;
  const int nb = blockIdx.x;
  const int mb = blockIdx.y;

  const v8f vzero = {0.f, 0.f, 0.f, 0.f, 0.f, 0.f, 0.f, 0.f};
  v8f acc[4][2];
#pragma unroll
  for (int mt = 0; mt < 4; ++mt) { acc[mt][0] = vzero; acc[mt][1] = vzero; }

  auto stage = [&](int buf, int k0) {
    // A: 128x32, 32B per thread, contiguous along K (async path)
    const int e = tid * 16;
    const int ar = e >> 5, ac = e & 31;
    const __bf16* gp = A + (size_t)(mb * 128 + ar) * Kdim + k0 + ac;
    g2l_16B(gp, &As[buf][ar][ac]);
    g2l_16B(gp + 8, &As[buf][ar][ac + 8]);
    // B transposed: read contiguous along N, scatter into Bts[n][k]
    const int br = e >> 7, bc = e & 127;
    const __bf16* gb = B + (size_t)(k0 + br) * Ndim + nb * 128 + bc;
    v8bf y0 = *(const v8bf*)gb;
    v8bf y1 = *(const v8bf*)(gb + 8);
#pragma unroll
    for (int j = 0; j < 8; ++j) {
      Bts[buf][bc + j][br]     = y0[j];
      Bts[buf][bc + 8 + j][br] = y1[j];
    }
  };

  const int nk = Kdim >> 5;
  stage(0, 0);
  async_join();
  __syncthreads();

  for (int kb = 0; kb < nk; ++kb) {
    const int cur = kb & 1;
    if (kb + 1 < nk) stage(cur ^ 1, (kb + 1) * 32);  // prefetch next tile

    v16bf bf0 = load_b_frag(&Bts[cur][wn * 32 + lx][0], 0, lane);
    v16bf bf1 = load_b_frag(&Bts[cur][wn * 32 + 16 + lx][0], 0, lane);
#pragma unroll
    for (int mt = 0; mt < 4; ++mt) {
      v16bf a = load_a_frag(&As[cur][wm * 64 + mt * 16 + lx][0], 0, lane);
      acc[mt][0] = wmma_bf16(a, bf0, acc[mt][0]);
      acc[mt][1] = wmma_bf16(a, bf1, acc[mt][1]);
    }
    async_join();
    __syncthreads();
  }

  const int typ = nb >> 4;  // 0=q 1=k 2=v 3=gate (MODE 0 only)
  const int h   = nb & 15;

  if constexpr (MODE == 0) {
    if (typ < 2) {
      // clamp + partial sum of squares per output row
      float part[4][8];
#pragma unroll
      for (int mt = 0; mt < 4; ++mt) {
#pragma unroll
        for (int i = 0; i < 8; ++i) {
          float v0 = fminf(fmaxf(acc[mt][0][i], -65504.f), 65504.f);
          float v1 = fminf(fmaxf(acc[mt][1][i], -65504.f), 65504.f);
          acc[mt][0][i] = v0; acc[mt][1][i] = v1;
          float p = v0 * v0 + v1 * v1;
#pragma unroll
          for (int off = 1; off < 16; off <<= 1) p += __shfl_xor(p, off, 16);
          part[mt][i] = p;
        }
      }
      if (lx == 0) {
#pragma unroll
        for (int mt = 0; mt < 4; ++mt)
#pragma unroll
          for (int i = 0; i < 8; ++i)
            red[wm * 64 + mt * 16 + hf * 8 + i][wn] = part[mt][i];
      }
      __syncthreads();
      __bf16* base = (typ == 0) ? Qo : Ko;
#pragma unroll
      for (int mt = 0; mt < 4; ++mt) {
#pragma unroll
        for (int i = 0; i < 8; ++i) {
          const int row = wm * 64 + mt * 16 + hf * 8 + i;
          float tot = red[row][0] + red[row][1] + red[row][2] + red[row][3];
          float r = rsqrtf(tot * (1.0f / 128.0f) + 1e-5f);
          const int mrow = mb * 128 + row;
          const int bb = mrow >> 11, s = mrow & 2047;
          __bf16* dst = base + ((size_t)(bb * 16 + h) * 2048 + s) * 128;
          dst[wn * 32 + lx]      = (__bf16)(acc[mt][0][i] * r);
          dst[wn * 32 + 16 + lx] = (__bf16)(acc[mt][1][i] * r);
        }
      }
    } else {
#pragma unroll
      for (int mt = 0; mt < 4; ++mt) {
#pragma unroll
        for (int i = 0; i < 8; ++i) {
          const int row = wm * 64 + mt * 16 + hf * 8 + i;
          const int mrow = mb * 128 + row;
          const int bb = mrow >> 11, s = mrow & 2047;
          if (typ == 2) {
            __bf16* dst = Vo + ((size_t)(bb * 16 + h) * 2048 + s) * 128;
            dst[wn * 32 + lx]      = (__bf16)acc[mt][0][i];
            dst[wn * 32 + 16 + lx] = (__bf16)acc[mt][1][i];
          } else {
            __bf16* dst = Go + (size_t)mrow * 2048 + h * 128;
            dst[wn * 32 + lx]      = (__bf16)(1.f / (1.f + __expf(-acc[mt][0][i])));
            dst[wn * 32 + 16 + lx] = (__bf16)(1.f / (1.f + __expf(-acc[mt][1][i])));
          }
        }
      }
    }
  } else {
#pragma unroll
    for (int mt = 0; mt < 4; ++mt) {
#pragma unroll
      for (int i = 0; i < 8; ++i) {
        const int row = wm * 64 + mt * 16 + hf * 8 + i;
        const int mrow = mb * 128 + row;
        float* dst = Cout + (size_t)mrow * Ndim + nb * 128;
        dst[wn * 32 + lx]      = acc[mt][0][i];
        dst[wn * 32 + 16 + lx] = acc[mt][1][i];
      }
    }
  }
}

// ---------------- Flash attention per (b, h, 128-query tile) ----------------
// 8 waves, each owns 16 query rows. Online softmax, WMMA for QK^T and PV.

__global__ __launch_bounds__(256) void attn_kernel(
    const __bf16* __restrict__ Q, const __bf16* __restrict__ Km,
    const __bf16* __restrict__ V, const __bf16* __restrict__ G,
    __bf16* __restrict__ AO) {
  __shared__ __bf16 Qs[128][128];   // [q][d]
  __shared__ __bf16 Ks[128][128];   // [key][d] -> B^T for QK^T
  __shared__ __bf16 Vts[128][128];  // [d][key] -> B^T for PV
  __shared__ __bf16 Ps[8][16][128]; // per-wave P tile [q][key]

  const int tid = threadIdx.x, lane = tid & 31, wave = tid >> 5;
  const int lx = lane & 15, hf = (lane >> 4) & 1;
  const int qt = blockIdx.x, h = blockIdx.y, bz = blockIdx.z;
  const size_t headoff = (size_t)(bz * 16 + h) * (2048 * 128);
  const __bf16* Qh = Q + headoff;
  const __bf16* Kh = Km + headoff;
  const __bf16* Vh = V + headoff;

  {  // stage Q tile (each thread: half a row = 64 bf16), async path
    const int r = tid >> 1, c0 = (tid & 1) * 64;
    const __bf16* src = Qh + (size_t)(qt * 128 + r) * 128 + c0;
#pragma unroll
    for (int j = 0; j < 8; ++j) g2l_16B(src + j * 8, &Qs[r][c0 + j * 8]);
  }

  const v8f vzero = {0.f, 0.f, 0.f, 0.f, 0.f, 0.f, 0.f, 0.f};
  v8f o[8];
  float mrun[8], srun[8];
#pragma unroll
  for (int i = 0; i < 8; ++i) { o[i] = vzero; mrun[i] = -3.0e38f; srun[i] = 0.f; }

  __bf16* Pw = &Ps[wave][0][0];
  const float scale = 0.08838834764831845f;  // 1/sqrt(128)

  for (int kt = 0; kt < 16; ++kt) {
    __syncthreads();
    {  // stage K (row-major, async) and V (transposed, manual)
      const int r = tid >> 1, c0 = (tid & 1) * 64;
      const __bf16* ks = Kh + (size_t)(kt * 128 + r) * 128 + c0;
#pragma unroll
      for (int j = 0; j < 8; ++j) g2l_16B(ks + j * 8, &Ks[r][c0 + j * 8]);
      const __bf16* vs = Vh + (size_t)(kt * 128 + r) * 128 + c0;
#pragma unroll
      for (int j = 0; j < 8; ++j) {
        v8bf vv = *(const v8bf*)(vs + j * 8);
#pragma unroll
        for (int u = 0; u < 8; ++u) Vts[c0 + j * 8 + u][r] = vv[u];
      }
    }
    async_join();
    __syncthreads();

    // S = Q * K^T  (16 q-rows x 128 keys per wave)
    v8f sacc[8];
#pragma unroll
    for (int nt = 0; nt < 8; ++nt) sacc[nt] = vzero;
#pragma unroll
    for (int kk = 0; kk < 4; ++kk) {
      v16bf a = load_a_frag(&Qs[wave * 16 + lx][0], kk * 32, lane);
#pragma unroll
      for (int nt = 0; nt < 8; ++nt) {
        v16bf b = load_b_frag(&Ks[nt * 16 + lx][0], kk * 32, lane);
        sacc[nt] = wmma_bf16(a, b, sacc[nt]);
      }
    }

    // online softmax (row stats across the 16-lane half holding the row)
#pragma unroll
    for (int i = 0; i < 8; ++i) {
      float mx = -3.0e38f;
#pragma unroll
      for (int nt = 0; nt < 8; ++nt) {
        float sv = sacc[nt][i] * scale;
        sacc[nt][i] = sv;
        mx = fmaxf(mx, sv);
      }
#pragma unroll
      for (int off = 1; off < 16; off <<= 1) mx = fmaxf(mx, __shfl_xor(mx, off, 16));
      const float mnew = fmaxf(mrun[i], mx);
      const float alpha = __expf(mrun[i] - mnew);
#pragma unroll
      for (int nt = 0; nt < 8; ++nt) o[nt][i] *= alpha;
      float ps = 0.f;
#pragma unroll
      for (int nt = 0; nt < 8; ++nt) {
        float p = __expf(sacc[nt][i] - mnew);
        ps += p;
        Pw[(size_t)(hf * 8 + i) * 128 + nt * 16 + lx] = (__bf16)p;
      }
#pragma unroll
      for (int off = 1; off < 16; off <<= 1) ps += __shfl_xor(ps, off, 16);
      srun[i] = srun[i] * alpha + ps;
      mrun[i] = mnew;
    }
    __syncthreads();

    // O += P * V
#pragma unroll
    for (int kk = 0; kk < 4; ++kk) {
      v16bf a = load_a_frag(Pw + (size_t)lx * 128, kk * 32, lane);
#pragma unroll
      for (int nt = 0; nt < 8; ++nt) {
        v16bf b = load_b_frag(&Vts[nt * 16 + lx][0], kk * 32, lane);
        o[nt] = wmma_bf16(a, b, o[nt]);
      }
    }
  }

  // finalize: 1/rowsum, sigmoid-gate, pack bf16 in [B*S, 2048] head-major layout
#pragma unroll
  for (int i = 0; i < 8; ++i) {
    const float inv = 1.0f / srun[i];
    const int qrow = qt * 128 + wave * 16 + hf * 8 + i;
    const size_t rowbase = ((size_t)bz * 2048 + qrow) * 2048 + h * 128;
#pragma unroll
    for (int nt = 0; nt < 8; ++nt) {
      const int d = nt * 16 + lx;
      const float g = (float)G[rowbase + d];
      AO[rowbase + d] = (__bf16)(o[nt][i] * inv * g);
    }
  }
}

// ---------------- host launch ----------------

extern "C" void kernel_launch(void* const* d_in, const int* in_sizes, int n_in,
                              void* d_out, int out_size, void* d_ws, size_t ws_size,
                              hipStream_t stream) {
  (void)in_sizes; (void)n_in; (void)out_size; (void)ws_size;
  const float* hs   = (const float*)d_in[0];  // [2,2048,2048]
  const float* win  = (const float*)d_in[1];  // [2048,8192]
  const float* wout = (const float*)d_in[2];  // [2048,2048]
  float* out = (float*)d_out;                 // [2,2048,2048]

  char* ws = (char*)d_ws;
  size_t off = 0;
  auto alloc = [&](size_t bytes) {
    void* p = ws + off;
    off += (bytes + 255) & ~(size_t)255;
    return p;
  };
  const size_t M = 4096;  // B*S
  __bf16* Xbf    = (__bf16*)alloc(M * 2048 * 2);        // hidden bf16
  __bf16* WinBf  = (__bf16*)alloc(2048 * 8192 * 2);     // W_in bf16
  __bf16* WoutBf = (__bf16*)alloc(2048 * 2048 * 2);     // W_out bf16
  __bf16* Qb     = (__bf16*)alloc(M * 2048 * 2);        // [B,H,S,128]
  __bf16* Kb     = (__bf16*)alloc(M * 2048 * 2);
  __bf16* Vb     = (__bf16*)alloc(M * 2048 * 2);
  __bf16* Gb     = (__bf16*)alloc(M * 2048 * 2);        // sigmoid gate, [M,2048]
  __bf16* AOb    = (__bf16*)alloc(M * 2048 * 2);        // gated attn out, [M,2048]

  cvt_f32_bf16<<<dim3(4096), dim3(256), 0, stream>>>(hs, Xbf, (int)(M * 2048));
  cvt_f32_bf16<<<dim3(4096), dim3(256), 0, stream>>>(win, WinBf, 2048 * 8192);
  cvt_f32_bf16<<<dim3(4096), dim3(256), 0, stream>>>(wout, WoutBf, 2048 * 2048);

  // proj + fused rmsnorm/sigmoid epilogue: N=8192 (64 n-blocks), M=4096 (32 m-blocks)
  gemm_bf16_kernel<0><<<dim3(64, 32), dim3(256), 0, stream>>>(
      Xbf, WinBf, 2048, 8192, Qb, Kb, Vb, Gb, nullptr);

  // attention: 16 q-tiles x 16 heads x 2 batches
  attn_kernel<<<dim3(16, 16, 2), dim3(256), 0, stream>>>(Qb, Kb, Vb, Gb, AOb);

  // output projection -> fp32 d_out
  gemm_bf16_kernel<1><<<dim3(16, 32), dim3(256), 0, stream>>>(
      AOb, WoutBf, 2048, 2048, nullptr, nullptr, nullptr, nullptr, out);
}
